// GPT_24635932410197
// MI455X (gfx1250) — compile-verified, matrix operand order checked by reading, not run on hardware
//
#include <hip/hip_runtime.h>

// ---------------- problem constants (GPT-2 small fwd) ----------------
#define BB 2
#define TT 1024
#define CC 768
#define LL 12
#define HH 12
#define VV 50257
#define DD 64               // head dim
#define MM (BB*TT)          // 2048 rows of activations

typedef unsigned short u16;
typedef __attribute__((ext_vector_type(16))) __bf16 v16bf;
typedef __attribute__((ext_vector_type(8)))  float  v8f;
typedef __attribute__((ext_vector_type(4)))  int    v4i;

union Frag { uint4 u[2]; v16bf v; };

__device__ __forceinline__ u16 f2bf(float f) {
  unsigned int u = __float_as_uint(f);
  u += 0x7FFFu + ((u >> 16) & 1u);   // round-to-nearest-even
  return (u16)(u >> 16);
}

__device__ __forceinline__ float gelu_f(float x) {
  const float k = 0.7978845608028654f;   // sqrt(2/pi)
  float x3 = x * x * x;
  return 0.5f * x * (1.0f + tanhf(k * (x + 0.044715f * x3)));
}

// ---------------- CDNA5 async global->LDS copy (ASYNCcnt path) ----------------
#if defined(__has_builtin)
#  if __has_builtin(__builtin_amdgcn_global_load_async_to_lds_b128)
#    define HAVE_ASYNC_BUILTIN 1
#  endif
#  if __has_builtin(__builtin_amdgcn_s_wait_asynccnt)
#    define HAVE_ASYNCWAIT_BUILTIN 1
#  endif
#endif

typedef __attribute__((address_space(1))) v4i* as1_v4i;
typedef __attribute__((address_space(3))) v4i* as3_v4i;

__device__ __forceinline__ void async_copy_b128(const u16* gsrc, u16* ldst) {
#if defined(HAVE_ASYNC_BUILTIN)
  __builtin_amdgcn_global_load_async_to_lds_b128(
      (as1_v4i)(size_t)gsrc, (as3_v4i)(unsigned)(size_t)ldst, 0, 0);
#else
  asm volatile("global_load_async_to_lds_b128 %0, %1, off"
               :: "v"((unsigned)(size_t)ldst),
                  "v"((unsigned long long)(size_t)gsrc)
               : "memory");
#endif
}

__device__ __forceinline__ void async_wait0() {
#if defined(HAVE_ASYNCWAIT_BUILTIN)
  __builtin_amdgcn_s_wait_asynccnt(0);
#else
  asm volatile("s_wait_asynccnt 0x0" ::: "memory");
#endif
}

// ---------------- weight conversion kernels ----------------
__global__ void k_convert(const float* __restrict__ in, u16* __restrict__ out, long n) {
  long i = (long)blockIdx.x * blockDim.x + threadIdx.x;
  long st = (long)gridDim.x * blockDim.x;
  for (; i < n; i += st) out[i] = f2bf(in[i]);
}

// per-layer transpose: in fp32 [layers][K][N] -> out bf16 [layers][N][K]
__global__ void k_convertT(const float* __restrict__ in, u16* __restrict__ out,
                           int K, int N, long total) {
  long i = (long)blockIdx.x * blockDim.x + threadIdx.x;
  long st = (long)gridDim.x * blockDim.x;
  long KN = (long)K * N;
  for (; i < total; i += st) {
    long l = i / KN, r = i - l * KN;
    int k = (int)(r / N), n = (int)(r - (long)k * N);
    out[l * KN + (long)n * K + k] = f2bf(in[i]);
  }
}

// ---------------- embedding ----------------
__global__ __launch_bounds__(256) void k_embed(const int* __restrict__ idx,
                                               const float* __restrict__ wte,
                                               const float* __restrict__ wpe,
                                               float* __restrict__ x) {
  int row = blockIdx.x;                 // 0..MM-1
  int t = row % TT;
  int tok = idx[row];
  const float* we = wte + (size_t)tok * CC;
  const float* wp = wpe + (size_t)t * CC;
  float* xr = x + (size_t)row * CC;
  for (int c = threadIdx.x; c < CC; c += 256) xr[c] = we[c] + wp[c];
}

// ---------------- layernorm (one block per row), bf16 output ----------------
__global__ __launch_bounds__(256) void k_layernorm(const float* __restrict__ x,
                                                   const float* __restrict__ w,
                                                   const float* __restrict__ b,
                                                   u16* __restrict__ out) {
  __shared__ float red[256];
  int r = blockIdx.x;
  const float* xr = x + (size_t)r * CC;
  float s = 0.f;
  for (int c = threadIdx.x; c < CC; c += 256) s += xr[c];
  red[threadIdx.x] = s; __syncthreads();
  for (int st = 128; st > 0; st >>= 1) {
    if (threadIdx.x < st) red[threadIdx.x] += red[threadIdx.x + st];
    __syncthreads();
  }
  float mean = red[0] * (1.0f / CC);
  __syncthreads();
  float v = 0.f;
  for (int c = threadIdx.x; c < CC; c += 256) { float d = xr[c] - mean; v += d * d; }
  red[threadIdx.x] = v; __syncthreads();
  for (int st = 128; st > 0; st >>= 1) {
    if (threadIdx.x < st) red[threadIdx.x] += red[threadIdx.x + st];
    __syncthreads();
  }
  float rstd = rsqrtf(red[0] * (1.0f / CC) + 1e-5f);
  u16* orow = out + (size_t)r * CC;
  for (int c = threadIdx.x; c < CC; c += 256)
    orow[c] = f2bf((xr[c] - mean) * rstd * w[c] + b[c]);
}

// ---------------- WMMA bf16 GEMM ----------------
// C[M,N] = Abf[M,K](bf16) * Bt[N,K](bf16) (+bias)(+epilogue)
// epi: 0 = bias only, 1 = bias+GELU, 2 = bias+residual add
// Cout (fp32) and/or Cbf (bf16) outputs, either may be null.
// block: 256 threads = 8 waves (4 M x 2 N); block tile 128(M) x 128(N) x 32(K)
// wave tile: 32(M) x 64(N) -> 2 A-frags x 4 B-frags = 8 WMMAs per K-step.
// Both tiles stream via global_load_async_to_lds_b128 (ASYNCcnt DMA path).
#define LDA 40   // row stride in u16: 80B rows keep frag offsets 16B aligned
__global__ __launch_bounds__(256) void k_gemm(const u16* __restrict__ Abf,
                                              const u16* __restrict__ Bt,
                                              const float* __restrict__ bias,
                                              const float* __restrict__ residual,
                                              float* __restrict__ Cout,
                                              u16* __restrict__ Cbf,
                                              int Ntot, int Ktot, int epi) {
  __shared__ __align__(16) u16 As[128 * LDA];
  __shared__ __align__(16) u16 Bs[128 * LDA];
  const int tid  = threadIdx.x;
  const int lane = tid & 31, wave = tid >> 5;
  const int lr   = lane & 15, half = lane >> 4;
  const int wm   = wave >> 1;     // 0..3 : 32-row M sub-tile
  const int wn   = wave & 1;      // 0..1 : 64-col N sub-tile
  const int mBase = blockIdx.y * 128;
  const int nBase = blockIdx.x * 128;
  const bool fullN = (nBase + 128 <= Ntot);

  v8f acc[2][4];
#pragma unroll
  for (int u = 0; u < 2; ++u)
#pragma unroll
    for (int t = 0; t < 4; ++t) acc[u][t] = (v8f){};

  for (int k0 = 0; k0 < Ktot; k0 += 32) {
    __syncthreads();
    if (k0 + 32 < Ktot) { // CDNA5 prefetch of next K-tile (global_prefetch_b8)
      __builtin_prefetch(&Abf[(size_t)(mBase + (tid >> 1)) * Ktot + k0 + 32], 0, 1);
      __builtin_prefetch(&Bt[(size_t)(nBase + (tid >> 1)) * Ktot + k0 + 32], 0, 1);
    }
    // ---- A tile: 128 rows x 32 K bf16, async DMA to LDS ----
#pragma unroll
    for (int i = 0; i < 2; ++i) {
      int s = i * 256 + tid;            // 512 segments of 8 elements
      int r = s >> 2, c = (s & 3) * 8;
      async_copy_b128(&Abf[(size_t)(mBase + r) * Ktot + k0 + c], &As[r * LDA + c]);
    }
    // ---- B tile: 128 rows x 32 K bf16, async DMA when row-range valid ----
    if (fullN) {
#pragma unroll
      for (int i = 0; i < 2; ++i) {
        int s = i * 256 + tid;
        int r = s >> 2, c = (s & 3) * 8;
        async_copy_b128(&Bt[(size_t)(nBase + r) * Ktot + k0 + c], &Bs[r * LDA + c]);
      }
    } else {                            // ragged lm-head edge tile only
#pragma unroll
      for (int i = 0; i < 2; ++i) {
        int s = i * 256 + tid;
        int r = s >> 2, c = (s & 3) * 8;
        int nn = nBase + r;
        uint4 v = make_uint4(0u, 0u, 0u, 0u);
        if (nn < Ntot) v = *(const uint4*)&Bt[(size_t)nn * Ktot + k0 + c];
        *(uint4*)&Bs[r * LDA + c] = v;
      }
    }
    async_wait0();
    __syncthreads();

    // ---- fragments (ISA 7.12.2 16-bit A layout; B mirrors with N<->M) ----
    Frag fa[2], fb[4];
#pragma unroll
    for (int u = 0; u < 2; ++u) {
      int m = wm * 32 + u * 16 + lr;
      fa[u].u[0] = *(const uint4*)&As[m * LDA + 8 * half];
      fa[u].u[1] = *(const uint4*)&As[m * LDA + 16 + 8 * half];
    }
#pragma unroll
    for (int t = 0; t < 4; ++t) {
      int n = wn * 64 + t * 16 + lr;
      fb[t].u[0] = *(const uint4*)&Bs[n * LDA + 8 * half];
      fb[t].u[1] = *(const uint4*)&Bs[n * LDA + 16 + 8 * half];
    }
#pragma unroll
    for (int u = 0; u < 2; ++u)
#pragma unroll
      for (int t = 0; t < 4; ++t)
        acc[u][t] = __builtin_amdgcn_wmma_f32_16x16x32_bf16(
            false, fa[u].v, false, fb[t].v, (short)0, acc[u][t], false, false);
  }

  // ---- epilogue: D f32 16x16 -> col = lr, element r -> row 8*half + r ----
#pragma unroll
  for (int u = 0; u < 2; ++u) {
#pragma unroll
    for (int t = 0; t < 4; ++t) {
      const float* ap = (const float*)&acc[u][t];
      int ncol = nBase + wn * 64 + t * 16 + lr;
      if (ncol < Ntot) {
        float bv = bias ? bias[ncol] : 0.f;
#pragma unroll
        for (int r = 0; r < 8; ++r) {
          int mrow = mBase + wm * 32 + u * 16 + half * 8 + r;
          float v = ap[r] + bv;
          if (epi == 1) v = gelu_f(v);
          size_t o = (size_t)mrow * Ntot + ncol;
          if (epi == 2) v += residual[o];
          if (Cout) Cout[o] = v;
          if (Cbf)  Cbf[o] = f2bf(v);
        }
      }
    }
  }
}

// ---------------- causal attention (flash-style, fp32 in, bf16 out) ----------------
__global__ __launch_bounds__(128) void k_attention(const float* __restrict__ qkv,
                                                   u16* __restrict__ y) {
  __shared__ float Ks[64][DD + 1];
  __shared__ float Vs[64][DD + 1];
  const int bh = blockIdx.x;
  const int b = bh / HH, h = bh % HH;
  const int tq = blockIdx.y * 128 + threadIdx.x;
  const size_t rowQ = (size_t)(b * TT + tq);
  float q[DD];
#pragma unroll
  for (int d = 0; d < DD; ++d) q[d] = qkv[rowQ * (3 * CC) + h * DD + d];
  float m = -INFINITY, l = 0.f, acc[DD];
#pragma unroll
  for (int d = 0; d < DD; ++d) acc[d] = 0.f;

  const int lastChunk = (blockIdx.y * 128 + 127) >> 6;
  for (int kc = 0; kc <= lastChunk; ++kc) {
    __syncthreads();
    for (int e = threadIdx.x; e < 64 * DD; e += 128) {
      int j = e / DD, d = e - j * DD;
      size_t rowK = (size_t)(b * TT + kc * 64 + j);
      Ks[j][d] = qkv[rowK * (3 * CC) + CC + h * DD + d];
      Vs[j][d] = qkv[rowK * (3 * CC) + 2 * CC + h * DD + d];
    }
    __syncthreads();
    for (int j = 0; j < 64; ++j) {
      int key = kc * 64 + j;
      if (key > tq) break;
      float s = 0.f;
#pragma unroll
      for (int d = 0; d < DD; ++d) s += q[d] * Ks[j][d];
      s *= 0.125f;
      float mn = fmaxf(m, s);
      float corr = expf(m - mn);
      float p = expf(s - mn);
      l = l * corr + p;
#pragma unroll
      for (int d = 0; d < DD; ++d) acc[d] = acc[d] * corr + p * Vs[j][d];
      m = mn;
    }
  }
  float inv = 1.f / l;
#pragma unroll
  for (int d = 0; d < DD; ++d) y[rowQ * CC + h * DD + d] = f2bf(acc[d] * inv);
}

// ---------------- loss ----------------
__global__ __launch_bounds__(256) void k_loss_row(const float* __restrict__ logits,
                                                  const int* __restrict__ targets,
                                                  float* __restrict__ nll) {
  __shared__ float red[256];
  int r = blockIdx.x;
  const float* lg = logits + (size_t)r * VV;
  float mx = -INFINITY;
  for (int v = threadIdx.x; v < VV; v += 256) mx = fmaxf(mx, lg[v]);
  red[threadIdx.x] = mx; __syncthreads();
  for (int st = 128; st > 0; st >>= 1) {
    if (threadIdx.x < st) red[threadIdx.x] = fmaxf(red[threadIdx.x], red[threadIdx.x + st]);
    __syncthreads();
  }
  float smax = red[0]; __syncthreads();
  float s = 0.f;
  for (int v = threadIdx.x; v < VV; v += 256) s += expf(lg[v] - smax);
  red[threadIdx.x] = s; __syncthreads();
  for (int st = 128; st > 0; st >>= 1) {
    if (threadIdx.x < st) red[threadIdx.x] += red[threadIdx.x + st];
    __syncthreads();
  }
  if (threadIdx.x == 0) {
    int t = targets[r];
    bool valid = (t != -1);
    int tt = valid ? t : 0;
    nll[r] = valid ? -(lg[tt] - smax - logf(red[0])) : 0.f;
  }
}

__global__ __launch_bounds__(256) void k_loss_final(const float* __restrict__ nll,
                                                    const int* __restrict__ targets,
                                                    float* __restrict__ out) {
  __shared__ float rs[256];
  __shared__ float rc[256];
  float s = 0.f, c = 0.f;
  for (int r = threadIdx.x; r < MM; r += 256) {
    s += nll[r];
    c += (targets[r] != -1) ? 1.f : 0.f;
  }
  rs[threadIdx.x] = s; rc[threadIdx.x] = c; __syncthreads();
  for (int st = 128; st > 0; st >>= 1) {
    if (threadIdx.x < st) { rs[threadIdx.x] += rs[threadIdx.x + st];
                            rc[threadIdx.x] += rc[threadIdx.x + st]; }
    __syncthreads();
  }
  if (threadIdx.x == 0) out[0] = rs[0] / fmaxf(rc[0], 1.f);
}

// ---------------- launch ----------------
extern "C" void kernel_launch(void* const* d_in, const int* in_sizes, int n_in,
                              void* d_out, int out_size, void* d_ws, size_t ws_size,
                              hipStream_t stream) {
  (void)in_sizes; (void)n_in; (void)out_size; (void)ws_size;
  const int*   idx      = (const int*)  d_in[0];
  const int*   targets  = (const int*)  d_in[1];
  const float* wte      = (const float*)d_in[2];
  const float* wpe      = (const float*)d_in[3];
  const float* ln1_w    = (const float*)d_in[4];
  const float* ln1_b    = (const float*)d_in[5];
  const float* attn_w   = (const float*)d_in[6];
  const float* attn_b   = (const float*)d_in[7];
  const float* aproj_w  = (const float*)d_in[8];
  const float* aproj_b  = (const float*)d_in[9];
  const float* ln2_w    = (const float*)d_in[10];
  const float* ln2_b    = (const float*)d_in[11];
  const float* fc_w     = (const float*)d_in[12];
  const float* fc_b     = (const float*)d_in[13];
  const float* mproj_w  = (const float*)d_in[14];
  const float* mproj_b  = (const float*)d_in[15];
  const float* lnf_w    = (const float*)d_in[16];
  const float* lnf_b    = (const float*)d_in[17];

  float* logits = (float*)d_out;
  float* lossp  = (float*)d_out + (size_t)MM * VV;

  char* ws = (char*)d_ws;
  size_t off = 0;
  auto carve = [&](size_t bytes) { char* p = ws + off; off = (off + bytes + 255) & ~(size_t)255; return p; };
  u16*   attn_wt  = (u16*)  carve((size_t)LL * 3 * CC * CC * 2);
  u16*   aproj_wt = (u16*)  carve((size_t)LL * CC * CC * 2);
  u16*   fc_wt    = (u16*)  carve((size_t)LL * 4 * CC * CC * 2);
  u16*   mproj_wt = (u16*)  carve((size_t)LL * 4 * CC * CC * 2);
  u16*   wte_bf   = (u16*)  carve((size_t)VV * CC * 2);
  float* x        = (float*)carve((size_t)MM * CC * 4);
  u16*   xn_bf    = (u16*)  carve((size_t)MM * CC * 2);
  float* qkv      = (float*)carve((size_t)MM * 3 * CC * 4);
  u16*   attny_bf = (u16*)  carve((size_t)MM * CC * 2);
  u16*   h4_bf    = (u16*)  carve((size_t)MM * 4 * CC * 2);
  float* nll      = (float*)carve((size_t)MM * 4);

  // ---- weight conversion (bf16, transposed to [N][K]) ----
  {
    long n;
    n = (long)LL * CC * 3 * CC;
    k_convertT<<<4096, 256, 0, stream>>>(attn_w,  attn_wt,  CC,     3 * CC, n);
    n = (long)LL * CC * CC;
    k_convertT<<<4096, 256, 0, stream>>>(aproj_w, aproj_wt, CC,     CC,     n);
    n = (long)LL * CC * 4 * CC;
    k_convertT<<<4096, 256, 0, stream>>>(fc_w,    fc_wt,    CC,     4 * CC, n);
    n = (long)LL * 4 * CC * CC;
    k_convertT<<<4096, 256, 0, stream>>>(mproj_w, mproj_wt, 4 * CC, CC,     n);
    n = (long)VV * CC;                  // wte [V][C] already [N][K] for lm-head
    k_convert<<<4096, 256, 0, stream>>>(wte, wte_bf, n);
  }

  // ---- embedding ----
  k_embed<<<MM, 256, 0, stream>>>(idx, wte, wpe, x);

  const int MT = MM / 128;   // 16 M-tiles
  for (int l = 0; l < LL; ++l) {
    k_layernorm<<<MM, 256, 0, stream>>>(x, ln1_w + (size_t)l * CC, ln1_b + (size_t)l * CC, xn_bf);
    k_gemm<<<dim3(3 * CC / 128, MT), 256, 0, stream>>>(
        xn_bf, attn_wt + (size_t)l * 3 * CC * CC, attn_b + (size_t)l * 3 * CC,
        nullptr, qkv, nullptr, 3 * CC, CC, 0);
    k_attention<<<dim3(BB * HH, TT / 128), 128, 0, stream>>>(qkv, attny_bf);
    k_gemm<<<dim3(CC / 128, MT), 256, 0, stream>>>(
        attny_bf, aproj_wt + (size_t)l * CC * CC, aproj_b + (size_t)l * CC,
        x, x, nullptr, CC, CC, 2);
    k_layernorm<<<MM, 256, 0, stream>>>(x, ln2_w + (size_t)l * CC, ln2_b + (size_t)l * CC, xn_bf);
    k_gemm<<<dim3(4 * CC / 128, MT), 256, 0, stream>>>(
        xn_bf, fc_wt + (size_t)l * 4 * CC * CC, fc_b + (size_t)l * 4 * CC,
        nullptr, nullptr, h4_bf, 4 * CC, CC, 1);
    k_gemm<<<dim3(CC / 128, MT), 256, 0, stream>>>(
        h4_bf, mproj_wt + (size_t)l * 4 * CC * CC, mproj_b + (size_t)l * CC,
        x, x, nullptr, CC, 4 * CC, 2);
  }

  k_layernorm<<<MM, 256, 0, stream>>>(x, lnf_w, lnf_b, xn_bf);
  k_gemm<<<dim3((VV + 127) / 128, MT), 256, 0, stream>>>(
      xn_bf, wte_bf, nullptr, nullptr, logits, nullptr, VV, CC, 0);
  k_loss_row<<<MM, 256, 0, stream>>>(logits, targets, nll);
  k_loss_final<<<1, 256, 0, stream>>>(nll, targets, lossp);
}
